// PositionAttention_721554505897
// MI455X (gfx1250) — matrix operand order repaired and checked by reading, not compiled, not run on hardware
//
#include <hip/hip_runtime.h>
#include <hip/hip_bf16.h>

// ---------------------------------------------------------------------------
// PositionAttention on gfx1250 (MI455X), wave32 + WMMA f16 (16x16x32, f32 acc)
//   Stage 1: Q/K = (W @ x)^T -> f16 (B,N,C);  V = W @ x -> f16 (B,C,N)
//   Stage 2: logits = Q K^T  -> f32 (N,N)  (per batch, ws reuse)
//   Stage 3: P = softmax(logits) -> f16 (N,N)
//   Stage 4: out = gamma * (P V^T)^T + x   (B tile reads V rows contiguously)
// Score/out GEMMs use GLOBAL_LOAD_ASYNC_TO_LDS_B128 with 2-deep LDS double
// buffering, waits via s_wait_asynccnt.  All staging loops use uniform
// (scalar) trip counts so no divergent exec-masked loops are generated.
// ---------------------------------------------------------------------------

typedef __attribute__((ext_vector_type(16))) _Float16 v16h;
typedef __attribute__((ext_vector_type(8)))  _Float16 v8h;
typedef __attribute__((ext_vector_type(4)))  _Float16 v4h;
typedef __attribute__((ext_vector_type(8)))  float    v8f;
typedef __attribute__((ext_vector_type(4)))  int      v4i;

#define NPIX 4096   // H*W
#define CDIM 512    // channels
#define BDIM 4      // batch

#define BM 128
#define BN 128
#define BK 32
#define LDA 40      // padded LDS row stride (halves)
#define LDB 40

// ----- async copy: global(16B) -> LDS, tracked by ASYNCcnt -----------------
#if defined(__has_builtin)
#if __has_builtin(__builtin_amdgcn_global_load_async_to_lds_b128)
#define HAVE_ASYNC_BUILTIN 1
#endif
#if __has_builtin(__builtin_amdgcn_s_wait_asynccnt)
#define HAVE_WAIT_BUILTIN 1
#endif
#endif
#ifndef HAVE_ASYNC_BUILTIN
#define HAVE_ASYNC_BUILTIN 0
#endif
#ifndef HAVE_WAIT_BUILTIN
#define HAVE_WAIT_BUILTIN 0
#endif

typedef __attribute__((address_space(1))) v4i glb_v4i;   // global int4
typedef __attribute__((address_space(3))) v4i lds_v4i;   // LDS int4

__device__ __forceinline__ void async_copy_b128(const _Float16* g, _Float16* l) {
#if HAVE_ASYNC_BUILTIN
  __builtin_amdgcn_global_load_async_to_lds_b128(
      (glb_v4i*)g, (lds_v4i*)l, 0, 0);
#else
  unsigned loff = (unsigned)(unsigned long long)(
      (__attribute__((address_space(3))) _Float16*)l);
  asm volatile("global_load_async_to_lds_b128 %0, %1, off"
               :: "v"(loff), "v"(g) : "memory");
#endif
}

#if HAVE_WAIT_BUILTIN
#define WAIT_ASYNC(n) __builtin_amdgcn_s_wait_asynccnt(n)
#else
#define WAIT_ASYNC(n) asm volatile("s_wait_asynccnt %0" :: "n"(n) : "memory")
#endif

// ----- WMMA fragment loads from LDS (layouts per CDNA5 ISA 7.12.2) ---------
__device__ __forceinline__ v16h frag_a(const _Float16* As, int mrow0) {
  const int lane = threadIdx.x & 31;
  const int row  = mrow0 + (lane & 15);
  const int klo  = (lane & 16) ? 8 : 0;
  union { v16h v; v8h h[2]; } u;
  u.h[0] = *(const v8h*)(As + row * LDA + klo);
  u.h[1] = *(const v8h*)(As + row * LDA + klo + 16);
  return u.v;
}
__device__ __forceinline__ v16h frag_b(const _Float16* Bs, int ncol0) {
  const int lane = threadIdx.x & 31;
  const int col  = ncol0 + (lane & 15);
  const int klo  = (lane & 16) ? 8 : 0;
  union { v16h v; v8h h[2]; } u;
  u.h[0] = *(const v8h*)(Bs + col * LDB + klo);
  u.h[1] = *(const v8h*)(Bs + col * LDB + klo + 16);
  return u.v;
}

// ---------------------------------------------------------------------------
// Stage 1: proj.  TRANS=1: Yt[n][o] (for Q,K).  TRANS=0: Yt[o][n] (for V).
// ---------------------------------------------------------------------------
template <bool TRANS>
__global__ __launch_bounds__(256)
void proj_gemm_kernel(const float* __restrict__ X, const float* __restrict__ W,
                      _Float16* __restrict__ Yt) {
  __shared__ _Float16 As[BM * LDA];
  __shared__ _Float16 Bs[BN * LDB];
  const int b  = blockIdx.z;
  const int n0 = blockIdx.x * BM;
  const int o0 = blockIdx.y * BN;
  const float* Xb = X + (size_t)b * CDIM * NPIX;
  _Float16*    Yb = Yt + (size_t)b * NPIX * CDIM;

  const int tid  = threadIdx.x;
  const int wave = tid >> 5;
  const int wm   = (wave >> 2) * 64;
  const int wn   = (wave & 3) * 32;

  v8f acc[4][2] = {};

  for (int k0 = 0; k0 < CDIM; k0 += BK) {
    // As[n][c] = X[c][n]   (uniform 16-iteration loop)
#pragma unroll
    for (int ii = 0; ii < (BM * BK) / 256; ++ii) {
      const int i = tid + ii * 256;
      const int c = i >> 7, n = i & 127;
      As[n * LDA + c] = (_Float16)Xb[(size_t)(k0 + c) * NPIX + (n0 + n)];
    }
    // Bs[o][c] = W[o][c]
#pragma unroll
    for (int ii = 0; ii < (BN * BK) / 256; ++ii) {
      const int i = tid + ii * 256;
      const int o = i >> 5, c = i & 31;
      Bs[o * LDB + c] = (_Float16)W[(size_t)(o0 + o) * CDIM + (k0 + c)];
    }
    __syncthreads();
    v16h a[4], bb[2];
    for (int mt = 0; mt < 4; mt++) a[mt]  = frag_a(As, wm + mt * 16);
    for (int nt = 0; nt < 2; nt++) bb[nt] = frag_b(Bs, wn + nt * 16);
    for (int mt = 0; mt < 4; mt++)
      for (int nt = 0; nt < 2; nt++)
        acc[mt][nt] = __builtin_amdgcn_wmma_f32_16x16x32_f16(
            false, a[mt], false, bb[nt], (short)0, acc[mt][nt], false, false);
    __syncthreads();
  }
  const int lane  = tid & 31;
  const int ncol  = lane & 15;
  const int mbase = (lane & 16) ? 8 : 0;
  for (int mt = 0; mt < 4; mt++)
    for (int nt = 0; nt < 2; nt++) {
      const int o  = o0 + wn + nt * 16 + ncol;
      const int nn = n0 + wm + mt * 16 + mbase;
      if (TRANS) {
        for (int r = 0; r < 8; r++)
          Yb[(size_t)(nn + r) * CDIM + o] = (_Float16)acc[mt][nt][r];
      } else {
        union { v8h v; _Float16 e[8]; } pk;
        for (int r = 0; r < 8; r++) pk.e[r] = (_Float16)acc[mt][nt][r];
        *(v8h*)&Yb[(size_t)o * NPIX + nn] = pk.v;   // 16B store, aligned
      }
    }
}

// ---------------------------------------------------------------------------
// Async tile stage: 128 rows x 32 halves, global row stride `gstride` halves,
// into LDS (padded stride `lstride`).  Fully unrolled: each thread issues
// exactly 2 b128 asyncs (rows tid>>2 and (tid>>2)+64), no divergent loop.
// ---------------------------------------------------------------------------
__device__ __forceinline__ void issue_tile_async(
    const _Float16* __restrict__ gbase, size_t gstride, int row0, int k0,
    _Float16* lds, int lstride) {
  const int tid = threadIdx.x;
  const int row = tid >> 2;
  const int q   = (tid & 3) * 8;
  async_copy_b128(gbase + (size_t)(row0 + row) * gstride + k0 + q,
                  lds + row * lstride + q);
  async_copy_b128(gbase + (size_t)(row0 + row + 64) * gstride + k0 + q,
                  lds + (row + 64) * lstride + q);
}

// ---------------------------------------------------------------------------
// Stage 2: L[n][m] = sum_c Q[n][c] * K[m][c]
// ---------------------------------------------------------------------------
__global__ __launch_bounds__(256)
void score_gemm_kernel(const _Float16* __restrict__ Q,
                       const _Float16* __restrict__ K,
                       float* __restrict__ L) {
  __shared__ _Float16 As[2][BM * LDA];
  __shared__ _Float16 Bs[2][BN * LDB];
  const int n0 = blockIdx.x * BM;
  const int m0 = blockIdx.y * BN;
  const int tid  = threadIdx.x;
  const int wave = tid >> 5;
  const int wm   = (wave >> 2) * 64;
  const int wn   = (wave & 3) * 32;

  v8f acc[4][2] = {};

  issue_tile_async(Q, CDIM, n0, 0, As[0], LDA);
  issue_tile_async(K, CDIM, m0, 0, Bs[0], LDB);

  const int NIT = CDIM / BK;
  for (int it = 0; it < NIT; ++it) {
    const int buf = it & 1;
    if (it + 1 < NIT) {
      issue_tile_async(Q, CDIM, n0, (it + 1) * BK, As[buf ^ 1], LDA);
      issue_tile_async(K, CDIM, m0, (it + 1) * BK, Bs[buf ^ 1], LDB);
      WAIT_ASYNC(4);          // previous tile's 4 asyncs done (per wave)
    } else {
      WAIT_ASYNC(0);
    }
    __syncthreads();
    v16h a[4], bb[2];
    for (int mt = 0; mt < 4; mt++) a[mt]  = frag_a(As[buf], wm + mt * 16);
    for (int nt = 0; nt < 2; nt++) bb[nt] = frag_b(Bs[buf], wn + nt * 16);
    for (int mt = 0; mt < 4; mt++)
      for (int nt = 0; nt < 2; nt++)
        acc[mt][nt] = __builtin_amdgcn_wmma_f32_16x16x32_f16(
            false, a[mt], false, bb[nt], (short)0, acc[mt][nt], false, false);
    __syncthreads();
  }
  const int lane  = tid & 31;
  const int ncol  = lane & 15;
  const int mbase = (lane & 16) ? 8 : 0;
  for (int mt = 0; mt < 4; mt++)
    for (int nt = 0; nt < 2; nt++) {
      const int m = m0 + wn + nt * 16 + ncol;
      for (int r = 0; r < 8; r++) {
        const int n = n0 + wm + mt * 16 + mbase + r;
        L[(size_t)n * NPIX + m] = acc[mt][nt][r];
      }
    }
}

// ---------------------------------------------------------------------------
// Stage 3: P[row][:] = softmax(L[row][:])
// ---------------------------------------------------------------------------
__global__ __launch_bounds__(256)
void softmax_kernel(const float* __restrict__ L, _Float16* __restrict__ P) {
  __shared__ float red[8];
  const int row = blockIdx.x;
  const int tid = threadIdx.x;
  const float4* rp4 = (const float4*)(L + (size_t)row * NPIX);
  _Float16*     pp  = P + (size_t)row * NPIX;

  float m = -3.4e38f;
#pragma unroll
  for (int ii = 0; ii < NPIX / 1024; ++ii) {
    float4 v = rp4[tid + ii * 256];
    m = fmaxf(m, fmaxf(fmaxf(v.x, v.y), fmaxf(v.z, v.w)));
  }
  for (int off = 16; off > 0; off >>= 1) m = fmaxf(m, __shfl_xor(m, off, 32));
  if ((tid & 31) == 0) red[tid >> 5] = m;
  __syncthreads();
  float mm = red[0];
  for (int i = 1; i < 8; i++) mm = fmaxf(mm, red[i]);
  __syncthreads();

  float s = 0.f;
#pragma unroll
  for (int ii = 0; ii < NPIX / 1024; ++ii) {
    float4 v = rp4[tid + ii * 256];
    s += __expf(v.x - mm) + __expf(v.y - mm) + __expf(v.z - mm) + __expf(v.w - mm);
  }
  for (int off = 16; off > 0; off >>= 1) s += __shfl_xor(s, off, 32);
  if ((tid & 31) == 0) red[tid >> 5] = s;
  __syncthreads();
  float ss = 0.f;
  for (int i = 0; i < 8; i++) ss += red[i];
  const float inv = 1.f / ss;

#pragma unroll
  for (int ii = 0; ii < NPIX / 1024; ++ii) {
    const int i = tid + ii * 256;
    float4 v = rp4[i];
    union { v4h v; _Float16 e[4]; } o;
    o.e[0] = (_Float16)(__expf(v.x - mm) * inv);
    o.e[1] = (_Float16)(__expf(v.y - mm) * inv);
    o.e[2] = (_Float16)(__expf(v.z - mm) * inv);
    o.e[3] = (_Float16)(__expf(v.w - mm) * inv);
    *(v4h*)&pp[i * 4] = o.v;   // 8B store
  }
}

// ---------------------------------------------------------------------------
// Stage 4: Out[o][n] = gamma * sum_m P[n][m] * V[o][m] + X[o][n]
//          (V stored (C, N): rows over channel o, contiguous in m)
// ---------------------------------------------------------------------------
__global__ __launch_bounds__(256)
void out_gemm_kernel(const _Float16* __restrict__ P,
                     const _Float16* __restrict__ V,
                     const float* __restrict__ X,
                     const float* __restrict__ gamma,
                     float* __restrict__ Out) {
  __shared__ _Float16 As[2][BM * LDA];
  __shared__ _Float16 Bs[2][BN * LDB];
  const int n0 = blockIdx.x * BM;   // over NPIX (rows of O)
  const int o0 = blockIdx.y * BN;   // over CDIM (cols of O)
  const int tid  = threadIdx.x;
  const int wave = tid >> 5;
  const int wm   = (wave >> 2) * 64;
  const int wn   = (wave & 3) * 32;

  v8f acc[4][2] = {};

  issue_tile_async(P, NPIX, n0, 0, As[0], LDA);
  issue_tile_async(V, NPIX, o0, 0, Bs[0], LDB);

  const int NIT = NPIX / BK;   // 128
  for (int it = 0; it < NIT; ++it) {
    const int buf = it & 1;
    if (it + 1 < NIT) {
      issue_tile_async(P, NPIX, n0, (it + 1) * BK, As[buf ^ 1], LDA);
      issue_tile_async(V, NPIX, o0, (it + 1) * BK, Bs[buf ^ 1], LDB);
      WAIT_ASYNC(4);
    } else {
      WAIT_ASYNC(0);
    }
    __syncthreads();
    v16h a[4], bb[2];
    for (int mt = 0; mt < 4; mt++) a[mt]  = frag_a(As[buf], wm + mt * 16);
    for (int nt = 0; nt < 2; nt++) bb[nt] = frag_b(Bs[buf], wn + nt * 16);
    for (int mt = 0; mt < 4; mt++)
      for (int nt = 0; nt < 2; nt++)
        acc[mt][nt] = __builtin_amdgcn_wmma_f32_16x16x32_f16(
            false, a[mt], false, bb[nt], (short)0, acc[mt][nt], false, false);
    __syncthreads();
  }
  const float g   = gamma[0];
  const int lane  = tid & 31;
  const int ncol  = lane & 15;
  const int mbase = (lane & 16) ? 8 : 0;
  for (int mt = 0; mt < 4; mt++)
    for (int nt = 0; nt < 2; nt++) {
      const int o  = o0 + wn + nt * 16 + ncol;
      const int nn = n0 + wm + mt * 16 + mbase;
      const size_t base = (size_t)o * NPIX + nn;   // 8 consecutive n per lane
      for (int r = 0; r < 8; r++)
        Out[base + r] = g * acc[mt][nt][r] + X[base + r];
    }
}

// ---------------------------------------------------------------------------
extern "C" void kernel_launch(void* const* d_in, const int* in_sizes, int n_in,
                              void* d_out, int out_size, void* d_ws, size_t ws_size,
                              hipStream_t stream) {
  const float* x     = (const float*)d_in[0];
  const float* Wb    = (const float*)d_in[1];
  const float* Wc    = (const float*)d_in[2];
  const float* Wd    = (const float*)d_in[3];
  const float* gamma = (const float*)d_in[4];
  float* out = (float*)d_out;

  // Workspace (144 MB): Qt/Kt (B,N,C) f16; Vt (B,C,N) f16; Lg (N,N) f32; P (N,N) f16
  char* ws = (char*)d_ws;
  _Float16* Qt = (_Float16*)ws;  ws += (size_t)BDIM * NPIX * CDIM * 2;
  _Float16* Kt = (_Float16*)ws;  ws += (size_t)BDIM * NPIX * CDIM * 2;
  _Float16* Vt = (_Float16*)ws;  ws += (size_t)BDIM * NPIX * CDIM * 2;
  float*    Lg = (float*)ws;     ws += (size_t)NPIX * NPIX * 4;
  _Float16* P  = (_Float16*)ws;

  const dim3 blk(256);
  const dim3 gproj(NPIX / BM, CDIM / BN, BDIM);
  proj_gemm_kernel<true><<<gproj, blk, 0, stream>>>(x, Wb, Qt);
  proj_gemm_kernel<true><<<gproj, blk, 0, stream>>>(x, Wc, Kt);
  proj_gemm_kernel<false><<<gproj, blk, 0, stream>>>(x, Wd, Vt);

  for (int b = 0; b < BDIM; b++) {
    const _Float16* Qb = Qt + (size_t)b * NPIX * CDIM;
    const _Float16* Kb = Kt + (size_t)b * NPIX * CDIM;
    const _Float16* Vb = Vt + (size_t)b * NPIX * CDIM;
    const float*    Xb = x   + (size_t)b * CDIM * NPIX;
    float*          Ob = out + (size_t)b * CDIM * NPIX;

    score_gemm_kernel<<<dim3(NPIX / BM, NPIX / BN), blk, 0, stream>>>(Qb, Kb, Lg);
    softmax_kernel<<<dim3(NPIX), blk, 0, stream>>>(Lg, P);
    out_gemm_kernel<<<dim3(NPIX / BM, CDIM / BN), blk, 0, stream>>>(P, Vb, Xb, gamma, Ob);
  }
}
